// NeuronLTX2TransformerBlock_65859028517588
// MI455X (gfx1250) — compile-verified, matrix operand order checked by reading, not run on hardware
//
#include <hip/hip_runtime.h>

// ---------------------------------------------------------------------------
// MI455X (gfx1250) implementation of the dual-stream transformer block.
// GEMMs run through v_wmma_f32_16x16x32_bf16 (wave32 WMMA), fp32 accumulate.
// Weights are transposed+converted to bf16 once per call so both WMMA
// operands are row-contiguous per lane. The block-shared B tile is staged
// into LDS with GLOBAL_LOAD_ASYNC_TO_LDS_B128 (ASYNCcnt) and double-buffered
// so the async DMA of step k+32 overlaps the WMMAs of step k; B fragments are
// then read with ds_load_b128 (XOR-swizzled to avoid bank conflicts).
// Running v/a states live in d_out; scratch is carved from d_ws.
// ---------------------------------------------------------------------------

typedef __attribute__((ext_vector_type(16))) __bf16 v16bf;
typedef __attribute__((ext_vector_type(8)))  float  v8f;
typedef unsigned short bfraw;                 // raw bf16 storage

struct alignas(16) U4 { unsigned x, y, z, w; };
union FragBF { v16bf v; U4 u[2]; };

__device__ __forceinline__ bfraw f2bf(float f) {
  unsigned u = __float_as_uint(f);
  u += 0x7FFFu + ((u >> 16) & 1u);            // round-to-nearest-even
  return (bfraw)(u >> 16);
}

// Async-stage one 64x32 bf16 B tile (4KB) into LDS: 256 threads x 16B.
// Chunk index is XOR-swizzled with row bits [3:2] so WMMA fragment reads
// (16 lanes, row stride 64B) land on distinct bank groups.
__device__ __forceinline__ void stage_B(const bfraw* __restrict__ Bt, int K,
                                        int n0, int k, char* bufp, int tid) {
  int rr = tid >> 2;                  // 0..63 : local output column
  int c  = tid & 3;                   // 16B chunk within the 64B row
  const bfraw* g = Bt + (size_t)(n0 + rr) * K + k + c * 8;
  unsigned lds = (unsigned)(size_t)(bufp + rr * 64 + ((c ^ ((rr >> 2) & 3)) << 4));
  asm volatile("global_load_async_to_lds_b128 %0, %1, off"
               :: "v"(lds), "v"(g) : "memory");
}

__device__ __forceinline__ void wait_async_all() {
  asm volatile("s_wait_asynccnt 0" ::: "memory");
}

// ---------------------------------------------------------------------------
// GEMM: C[M,N] = A[M,K] @ Bt[N,K]^T (+bias). A,Bt bf16 row-major, C fp32.
// Block = 256 threads (8 waves). Wave w -> rows [blockIdx.y*128 + w*16, +16),
// cols [blockIdx.x*64, +64) as 4 WMMA tiles sharing the A fragment; the B
// tile is shared block-wide via async-LDS double buffering.
// Requires M%128==0, N%64==0, K%32==0 (true for every GEMM in this model).
// ---------------------------------------------------------------------------
__global__ __launch_bounds__(256)
void k_gemm_bf16(const bfraw* __restrict__ A, const bfraw* __restrict__ Bt,
                 const float* __restrict__ bias, float* __restrict__ C,
                 int M, int N, int K) {
  __shared__ char Bsm[2][64 * 64];    // 2 x 4KB B tiles
  int tid  = threadIdx.x;
  int wave = tid >> 5;
  int lane = tid & 31;
  int m0 = blockIdx.y * 128 + wave * 16;
  int n0 = blockIdx.x * 64;
  int r  = lane & 15;                 // matrix row (A) / col (B) within tile
  int kh = (lane >> 4) << 3;          // K sub-offset: 0 for lanes 0-15, 8 else
  int c0 = kh >> 3;                   // first 16B chunk index (0 or 1)
  int sw = 0;                         // per-j swizzle recomputed below

  const bfraw* ap = A + (size_t)(m0 + r) * K + kh;

  v8f z = {0.f,0.f,0.f,0.f,0.f,0.f,0.f,0.f};
  v8f acc[4] = {z, z, z, z};

  stage_B(Bt, K, n0, 0, Bsm[0], tid);
  wait_async_all();
  __syncthreads();

  int buf = 0;
  for (int k = 0; k < K; k += 32) {
    if (k + 32 < K) stage_B(Bt, K, n0, k + 32, Bsm[buf ^ 1], tid);
    __builtin_prefetch(ap + k + 256, 0, 1);   // global_prefetch_b8 (A stream)

    FragBF a;
    a.u[0] = *(const U4*)(ap + k);            // K = k+kh .. k+kh+7
    a.u[1] = *(const U4*)(ap + k + 16);       // K = k+kh+16 .. k+kh+23

#pragma unroll
    for (int j = 0; j < 4; ++j) {
      int rr = j * 16 + r;                    // local B column for this tile
      sw = (rr >> 2) & 3;
      const char* base = &Bsm[buf][rr * 64];
      FragBF b;
      b.u[0] = *(const U4*)(base + ((c0 ^ sw) << 4));         // K kh..kh+7
      b.u[1] = *(const U4*)(base + (((c0 + 2) ^ sw) << 4));   // K kh+16..kh+23
      acc[j] = __builtin_amdgcn_wmma_f32_16x16x32_bf16(
          false, a.v, false, b.v, (short)0, acc[j], false, false);
    }
    wait_async_all();     // own async stage (next buffer) complete
    __syncthreads();      // all waves: stages done + reads of old buffer done
    buf ^= 1;
  }

  // C/D layout: lane holds col n = (lane&15); VGPR v holds row v + 8*(lane>>4)
  int cn  = n0 + r;
  int cmb = m0 + ((lane >> 4) << 3);
#pragma unroll
  for (int j = 0; j < 4; ++j) {
    float bv = bias ? bias[cn + 16 * j] : 0.0f;
#pragma unroll
    for (int vv = 0; vv < 8; ++vv)
      C[(size_t)(cmb + vv) * N + (cn + 16 * j)] = acc[j][vv] + bv;
  }
}

// ---------------------------------------------------------------------------
// Tiled transpose + fp32->bf16 convert. in: [R,Cc] fp32 (row stride rstride),
// out: [Cc,R] bf16 contiguous. blockIdx.z batches (per-head v transpose).
// ---------------------------------------------------------------------------
__global__ __launch_bounds__(256)
void k_transpose_cvt(const float* __restrict__ in, bfraw* __restrict__ out,
                     int R, int Cc, long rstride, long in_bstride, long out_bstride) {
  __shared__ float tile[32][33];
  in  += (size_t)blockIdx.z * in_bstride;
  out += (size_t)blockIdx.z * out_bstride;
  int r0 = blockIdx.y * 32, c0 = blockIdx.x * 32;
  int tx = threadIdx.x & 31, ty = threadIdx.x >> 5;   // 8 rows per pass
#pragma unroll
  for (int i = 0; i < 32; i += 8) {
    int r = r0 + ty + i, c = c0 + tx;
    tile[ty + i][tx] = (r < R && c < Cc) ? in[(size_t)r * rstride + c] : 0.0f;
  }
  __syncthreads();
#pragma unroll
  for (int i = 0; i < 32; i += 8) {
    int oc = c0 + ty + i;    // out row = original col
    int orr = r0 + tx;       // out col = original row
    if (oc < Cc && orr < R)
      out[(size_t)oc * R + orr] = f2bf(tile[tx][ty + i]);
  }
}

// AdaLN modulate (optional) + fp32->bf16 convert, elementwise over [S,D].
__global__ void k_modulate_cvt(const float* __restrict__ x,
                               const float* __restrict__ sh,
                               const float* __restrict__ sc,
                               bfraw* __restrict__ out, size_t n, int D) {
  size_t i = (size_t)blockIdx.x * 256 + threadIdx.x;
  if (i >= n) return;
  float v = x[i];
  if (sh) { int d = (int)(i % (size_t)D); v = v * (1.0f + sc[d]) + sh[d]; }
  out[i] = f2bf(v);
}

// Per-head RMSNorm over head_dim=64; [S,H*64] fp32 -> [H][S][64] bf16.
__global__ __launch_bounds__(32)
void k_rmsnorm_heads(const float* __restrict__ Y, const float* __restrict__ w,
                     bfraw* __restrict__ out, int S, int H) {
  int s = blockIdx.x, h = blockIdx.y, lane = threadIdx.x;
  const float* row = Y + ((size_t)s * H + h) * 64;
  float x0 = row[lane], x1 = row[lane + 32];
  float ss = x0 * x0 + x1 * x1;
#pragma unroll
  for (int m = 16; m > 0; m >>= 1) ss += __shfl_xor(ss, m, 32);
  float inv = rsqrtf(ss * (1.0f / 64.0f) + 1e-6f);
  bfraw* o = out + ((size_t)h * S + s) * 64;
  o[lane]      = f2bf(w[lane]      * x0 * inv);
  o[lane + 32] = f2bf(w[lane + 32] * x1 * inv);
}

// Row softmax with scale; fp32 [S,Mv] -> bf16 [S,Mv]. One block per row.
__global__ __launch_bounds__(256)
void k_softmax_cvt(const float* __restrict__ in, bfraw* __restrict__ out,
                   int Mv, float scale) {
  int s = blockIdx.x;
  const float* row = in + (size_t)s * Mv;
  bfraw* orow = out + (size_t)s * Mv;
  __shared__ float red[8];
  int t = threadIdx.x, lane = t & 31, w = t >> 5;
  float mx = -3.402823466e38f;
  for (int j = t; j < Mv; j += 256) mx = fmaxf(mx, row[j] * scale);
#pragma unroll
  for (int m = 16; m > 0; m >>= 1) mx = fmaxf(mx, __shfl_xor(mx, m, 32));
  if (lane == 0) red[w] = mx;
  __syncthreads();
  float gmax = red[0];
#pragma unroll
  for (int i = 1; i < 8; ++i) gmax = fmaxf(gmax, red[i]);
  __syncthreads();
  float sum = 0.f;
  for (int j = t; j < Mv; j += 256) sum += __expf(row[j] * scale - gmax);
#pragma unroll
  for (int m = 16; m > 0; m >>= 1) sum += __shfl_xor(sum, m, 32);
  if (lane == 0) red[w] = sum;
  __syncthreads();
  float gsum = 0.f;
#pragma unroll
  for (int i = 0; i < 8; ++i) gsum += red[i];
  float rinv = 1.0f / gsum;
  for (int j = t; j < Mv; j += 256)
    orow[j] = f2bf(__expf(row[j] * scale - gmax) * rinv);
}

// [H][S][64] fp32 -> [S, H*64] bf16
__global__ void k_merge_heads(const float* __restrict__ OH, bfraw* __restrict__ out,
                              int S, int H) {
  size_t n = (size_t)S * H * 64;
  size_t i = (size_t)blockIdx.x * 256 + threadIdx.x;
  if (i >= n) return;
  int d = (int)(i & 63);
  size_t sh = i >> 6;
  int h = (int)(sh % (size_t)H);
  size_t s = sh / (size_t)H;
  out[i] = f2bf(OH[((size_t)h * S + s) * 64 + d]);
}

// Gated GELU (tanh approx): Y [S,2F] fp32 -> out [S,F] bf16
__global__ void k_glu_cvt(const float* __restrict__ Y, bfraw* __restrict__ out,
                          int S, int F) {
  size_t n = (size_t)S * F;
  size_t i = (size_t)blockIdx.x * 256 + threadIdx.x;
  if (i >= n) return;
  size_t s = i / (size_t)F;
  int j = (int)(i % (size_t)F);
  float h = Y[s * (size_t)(2 * F) + j];
  float g = Y[s * (size_t)(2 * F) + F + j];
  float tin = 0.7978845608028654f * (g + 0.044715f * g * g * g);
  float gl = 0.5f * g * (1.0f + tanhf(tin));
  out[i] = f2bf(h * gl);
}

// dst += (gate ? gate[d] : 1) * src, elementwise over [S,D]
__global__ void k_residual_add(float* __restrict__ dst, const float* __restrict__ src,
                               const float* __restrict__ gate, size_t n, int D) {
  size_t i = (size_t)blockIdx.x * 256 + threadIdx.x;
  if (i >= n) return;
  float g = gate ? gate[i % (size_t)D] : 1.0f;
  dst[i] += g * src[i];
}

__global__ void k_add_rows(const float* __restrict__ a, const float* __restrict__ b,
                           float* __restrict__ o, int n) {
  int i = blockIdx.x * 256 + threadIdx.x;
  if (i < n) o[i] = a[i] + b[i];
}

// ---------------------------------------------------------------------------
// Host-side orchestration
// ---------------------------------------------------------------------------
static inline unsigned cdiv(size_t a, size_t b) { return (unsigned)((a + b - 1) / b); }

struct Ws {
  bfraw *WT, *XB, *XBE, *QH, *KH, *VT, *WSM, *OB;
  float *Y, *SC, *OH, *MODV, *MODA;
};

// W [Kd,Nd] fp32 -> WT bf16 transposed, then Y[Sr,Nd] = Xb @ W + bias
static void linear(hipStream_t st, const bfraw* Xb, int Sr, int Kd,
                   const float* W, const float* bias, int Nd,
                   bfraw* WT, float* Yout) {
  k_transpose_cvt<<<dim3(cdiv(Nd, 32), cdiv(Kd, 32), 1), 256, 0, st>>>(
      W, WT, Kd, Nd, (long)Nd, 0L, 0L);
  k_gemm_bf16<<<dim3((unsigned)(Nd / 64), (unsigned)(Sr / 128), 1), 256, 0, st>>>(
      Xb, WT, bias, Yout, Sr, Nd, Kd);
}

static void attention(hipStream_t st, Ws& ws, float* resid, int S, int Dq, int H,
                      const bfraw* Xb, const bfraw* Eb, int Mkv, int Dk,
                      const float* wq, const float* bq, const float* wk, const float* bk,
                      const float* wv, const float* bv, const float* wo, const float* bo,
                      const float* nq, const float* nk, const float* gate) {
  int inner = H * 64;
  // Q/K projections + per-head RMSNorm into head-major layouts
  linear(st, Xb, S, Dq, wq, bq, inner, ws.WT, ws.Y);
  k_rmsnorm_heads<<<dim3((unsigned)S, (unsigned)H), 32, 0, st>>>(ws.Y, nq, ws.QH, S, H);
  linear(st, Eb, Mkv, Dk, wk, bk, inner, ws.WT, ws.Y);
  k_rmsnorm_heads<<<dim3((unsigned)Mkv, (unsigned)H), 32, 0, st>>>(ws.Y, nk, ws.KH, Mkv, H);
  // V projection, then per-head transpose -> VT[h] = v_h^T [64, Mkv] bf16
  linear(st, Eb, Mkv, Dk, wv, bv, inner, ws.WT, ws.Y);
  k_transpose_cvt<<<dim3(2, cdiv(Mkv, 32), (unsigned)H), 256, 0, st>>>(
      ws.Y, ws.VT, Mkv, 64, (long)inner, 64L, (long)64 * Mkv);
  // Per-head: scores -> softmax -> o
  for (int h = 0; h < H; ++h) {
    const bfraw* qh = ws.QH + (size_t)h * S * 64;
    const bfraw* kh = ws.KH + (size_t)h * Mkv * 64;
    const bfraw* vt = ws.VT + (size_t)h * 64 * Mkv;
    k_gemm_bf16<<<dim3((unsigned)(Mkv / 64), (unsigned)(S / 128), 1), 256, 0, st>>>(
        qh, kh, nullptr, ws.SC, S, Mkv, 64);
    k_softmax_cvt<<<(unsigned)S, 256, 0, st>>>(ws.SC, ws.WSM, Mkv, 0.125f);
    k_gemm_bf16<<<dim3(1, (unsigned)(S / 128), 1), 256, 0, st>>>(
        ws.WSM, vt, nullptr, ws.OH + (size_t)h * S * 64, S, 64, Mkv);
  }
  size_t no = (size_t)S * inner;
  k_merge_heads<<<cdiv(no, 256), 256, 0, st>>>(ws.OH, ws.OB, S, H);
  linear(st, ws.OB, S, inner, wo, bo, Dq, ws.WT, ws.Y);
  size_t n = (size_t)S * Dq;
  k_residual_add<<<cdiv(n, 256), 256, 0, st>>>(resid, ws.Y, gate, n, Dq);
}

static void ffn(hipStream_t st, Ws& ws, float* resid, int S, int D, int F,
                const float* w1, const float* b1, const float* w2, const float* b2,
                const float* shm, const float* scm, const float* gm) {
  size_t n = (size_t)S * D;
  k_modulate_cvt<<<cdiv(n, 256), 256, 0, st>>>(resid, shm, scm, ws.XB, n, D);
  linear(st, ws.XB, S, D, w1, b1, 2 * F, ws.WT, ws.Y);
  size_t ng = (size_t)S * F;
  k_glu_cvt<<<cdiv(ng, 256), 256, 0, st>>>(ws.Y, ws.XB, S, F);
  linear(st, ws.XB, S, F, w2, b2, D, ws.WT, ws.Y);
  k_residual_add<<<cdiv(n, 256), 256, 0, st>>>(resid, ws.Y, gm, n, D);
}

extern "C" void kernel_launch(void* const* d_in, const int* in_sizes, int n_in,
                              void* d_out, int out_size, void* d_ws, size_t ws_size,
                              hipStream_t stream) {
  (void)in_sizes; (void)n_in; (void)out_size; (void)ws_size;
  const int NV = 2048, NA = 512, MV = 256, MA = 128;
  const int DV = 2048, DA = 1024, HV = 32, HA = 16;
  const int FFV = 4096, FFA = 2048;

  auto F32 = [&](int i) { return (const float*)d_in[i]; };
  // d_in order: inputs 0..5, then params flattened in setup_inputs() insertion
  // order. Each attn dict: wq,bq,wk,bk,wv,bv,wo,bo,nq,nk.
  const float* v_in = F32(0);
  const float* a_in = F32(1);
  const float* v_enc = F32(2);
  const float* a_enc = F32(3);
  const float* v_temb = F32(4);
  const float* a_temb = F32(5);
  const int ATTN1 = 6, ATTN2 = 16, AATTN1 = 26, AATTN2 = 36, A2V = 46, V2A = 56;
  const int FFI = 66, AFFI = 70, SST = 74, ASST = 75;

  // --- workspace carve (256B aligned bump allocator) ---
  char* wp = (char*)d_ws;
  auto alloc = [&](size_t bytes) -> char* {
    char* p = wp; wp += (bytes + 255) & ~(size_t)255; return p;
  };
  Ws ws;
  ws.MODV = (float*)alloc((size_t)6 * DV * 4);
  ws.MODA = (float*)alloc((size_t)6 * DA * 4);
  ws.WT  = (bfraw*)alloc((size_t)8192 * 2048 * 2);
  ws.XB  = (bfraw*)alloc((size_t)2048 * 4096 * 2);
  ws.XBE = (bfraw*)alloc((size_t)2048 * 2048 * 2);
  ws.Y   = (float*)alloc((size_t)2048 * 8192 * 4);
  ws.QH  = (bfraw*)alloc((size_t)2048 * 2048 * 2);
  ws.KH  = (bfraw*)alloc((size_t)2048 * 2048 * 2);
  ws.VT  = (bfraw*)alloc((size_t)2048 * 2048 * 2);
  ws.SC  = (float*)alloc((size_t)2048 * 2048 * 4);
  ws.WSM = (bfraw*)alloc((size_t)2048 * 2048 * 2);
  ws.OH  = (float*)alloc((size_t)2048 * 2048 * 4);
  ws.OB  = (bfraw*)alloc((size_t)2048 * 2048 * 2);

  // running states live in d_out: [v | a]
  float* v = (float*)d_out;
  float* a = v + (size_t)NV * DV;
  hipMemcpyAsync(v, v_in, (size_t)NV * DV * 4, hipMemcpyDeviceToDevice, stream);
  hipMemcpyAsync(a, a_in, (size_t)NA * DA * 4, hipMemcpyDeviceToDevice, stream);

  // AdaLN rows: sst + temb -> [6, D]; order: sh, sc, g, shm, scm, gm
  k_add_rows<<<cdiv((size_t)6 * DV, 256), 256, 0, stream>>>(F32(SST), v_temb, ws.MODV, 6 * DV);
  k_add_rows<<<cdiv((size_t)6 * DA, 256), 256, 0, stream>>>(F32(ASST), a_temb, ws.MODA, 6 * DA);
  const float *v_sh = ws.MODV, *v_sc = ws.MODV + DV, *v_g = ws.MODV + 2 * DV;
  const float *v_shm = ws.MODV + 3 * DV, *v_scm = ws.MODV + 4 * DV, *v_gm = ws.MODV + 5 * DV;
  const float *a_sh = ws.MODA, *a_sc = ws.MODA + DA, *a_g = ws.MODA + 2 * DA;
  const float *a_shm = ws.MODA + 3 * DA, *a_scm = ws.MODA + 4 * DA, *a_gm = ws.MODA + 5 * DA;

  auto AP = [&](int base, int j) { return F32(base + j); }; // wq..nk = 0..9

  // 1) video self-attn, AdaLN-modulated input, gated residual
  k_modulate_cvt<<<cdiv((size_t)NV * DV, 256), 256, 0, stream>>>(
      v, v_sh, v_sc, ws.XB, (size_t)NV * DV, DV);
  attention(stream, ws, v, NV, DV, HV, ws.XB, ws.XB, NV, DV,
            AP(ATTN1,0),AP(ATTN1,1),AP(ATTN1,2),AP(ATTN1,3),AP(ATTN1,4),
            AP(ATTN1,5),AP(ATTN1,6),AP(ATTN1,7),AP(ATTN1,8),AP(ATTN1,9), v_g);
  // 2) video cross-attn to text
  k_modulate_cvt<<<cdiv((size_t)NV * DV, 256), 256, 0, stream>>>(
      v, nullptr, nullptr, ws.XB, (size_t)NV * DV, DV);
  k_modulate_cvt<<<cdiv((size_t)MV * DV, 256), 256, 0, stream>>>(
      v_enc, nullptr, nullptr, ws.XBE, (size_t)MV * DV, DV);
  attention(stream, ws, v, NV, DV, HV, ws.XB, ws.XBE, MV, DV,
            AP(ATTN2,0),AP(ATTN2,1),AP(ATTN2,2),AP(ATTN2,3),AP(ATTN2,4),
            AP(ATTN2,5),AP(ATTN2,6),AP(ATTN2,7),AP(ATTN2,8),AP(ATTN2,9), nullptr);
  // 3) audio self-attn
  k_modulate_cvt<<<cdiv((size_t)NA * DA, 256), 256, 0, stream>>>(
      a, a_sh, a_sc, ws.XB, (size_t)NA * DA, DA);
  attention(stream, ws, a, NA, DA, HA, ws.XB, ws.XB, NA, DA,
            AP(AATTN1,0),AP(AATTN1,1),AP(AATTN1,2),AP(AATTN1,3),AP(AATTN1,4),
            AP(AATTN1,5),AP(AATTN1,6),AP(AATTN1,7),AP(AATTN1,8),AP(AATTN1,9), a_g);
  // 4) audio cross-attn to text
  k_modulate_cvt<<<cdiv((size_t)NA * DA, 256), 256, 0, stream>>>(
      a, nullptr, nullptr, ws.XB, (size_t)NA * DA, DA);
  k_modulate_cvt<<<cdiv((size_t)MA * DA, 256), 256, 0, stream>>>(
      a_enc, nullptr, nullptr, ws.XBE, (size_t)MA * DA, DA);
  attention(stream, ws, a, NA, DA, HA, ws.XB, ws.XBE, MA, DA,
            AP(AATTN2,0),AP(AATTN2,1),AP(AATTN2,2),AP(AATTN2,3),AP(AATTN2,4),
            AP(AATTN2,5),AP(AATTN2,6),AP(AATTN2,7),AP(AATTN2,8),AP(AATTN2,9), nullptr);
  // 5) audio -> video cross attention (video queries, audio kv)
  k_modulate_cvt<<<cdiv((size_t)NV * DV, 256), 256, 0, stream>>>(
      v, nullptr, nullptr, ws.XB, (size_t)NV * DV, DV);
  k_modulate_cvt<<<cdiv((size_t)NA * DA, 256), 256, 0, stream>>>(
      a, nullptr, nullptr, ws.XBE, (size_t)NA * DA, DA);
  attention(stream, ws, v, NV, DV, HA, ws.XB, ws.XBE, NA, DA,
            AP(A2V,0),AP(A2V,1),AP(A2V,2),AP(A2V,3),AP(A2V,4),
            AP(A2V,5),AP(A2V,6),AP(A2V,7),AP(A2V,8),AP(A2V,9), nullptr);
  // 6) video -> audio cross attention (audio queries, video kv)
  k_modulate_cvt<<<cdiv((size_t)NA * DA, 256), 256, 0, stream>>>(
      a, nullptr, nullptr, ws.XB, (size_t)NA * DA, DA);
  k_modulate_cvt<<<cdiv((size_t)NV * DV, 256), 256, 0, stream>>>(
      v, nullptr, nullptr, ws.XBE, (size_t)NV * DV, DV);
  attention(stream, ws, a, NA, DA, HA, ws.XB, ws.XBE, NV, DV,
            AP(V2A,0),AP(V2A,1),AP(V2A,2),AP(V2A,3),AP(V2A,4),
            AP(V2A,5),AP(V2A,6),AP(V2A,7),AP(V2A,8),AP(V2A,9), nullptr);
  // 7) gated FFNs with AdaLN modulation
  ffn(stream, ws, v, NV, DV, FFV, F32(FFI), F32(FFI+1), F32(FFI+2), F32(FFI+3),
      v_shm, v_scm, v_gm);
  ffn(stream, ws, a, NA, DA, FFA, F32(AFFI), F32(AFFI+1), F32(AFFI+2), F32(AFFI+3),
      a_shm, a_scm, a_gm);
}
